// NERRelationModel_84963043050124
// MI455X (gfx1250) — matrix operand order repaired and checked by reading, not compiled, not run on hardware
//
#include <hip/hip_runtime.h>
#include <cstdint>
#include <cstddef>

// ---------------------------------------------------------------------------
// CDNA5 (gfx1250) implementation of the NER+Relation model.
// Big matmuls use v_wmma_f32_16x16x32_bf16 (wave32, 16x16 tile per wave,
// f32 accumulate). B operands are pre-packed into WMMA fragment layout.
// ---------------------------------------------------------------------------

typedef __attribute__((ext_vector_type(16))) __bf16 v16bf;
typedef __attribute__((ext_vector_type(8)))  float  v8f;

namespace {
constexpr int cB = 64, cS = 512, cH = 768, cE = 32, cP = 16, cK = 9, cR = 6;
}

// ---------------- pack f32 [K,N] weights -> bf16 WMMA B fragments ----------
// fragment layout: elem index = ((kt*NT + nt)*32 + lane)*16 + e
//   k = kt*32 + (lane>>4)*8 + e + (e>=8 ? 8 : 0) ; n = nt*16 + (lane&15) ; zero padded OOB.
__global__ __launch_bounds__(256) void pack_b_kernel(
    const float* __restrict__ W, unsigned short* __restrict__ Bp,
    int K, int N, int NT)
{
  int idx = blockIdx.x * 256 + threadIdx.x;
  int KT = (K + 31) >> 5;
  if (idx >= KT * NT * 32) return;
  int lane = idx & 31;
  int tile = idx >> 5;
  int nt = tile % NT;
  int kt = tile / NT;
  int n  = nt * 16 + (lane & 15);
  int kb = kt * 32 + (lane >> 4) * 8;
  __bf16* dst = (__bf16*)(Bp + (size_t)idx * 16);
#pragma unroll
  for (int e = 0; e < 16; ++e) {
    int k = kb + e + ((e >= 8) ? 8 : 0);
    float v = (k < K && n < N) ? W[(size_t)k * N + n] : 0.0f;
    dst[e] = (__bf16)v;
  }
}

// ---------------- generic WMMA GEMM: C = act(A @ B + bias) -----------------
// A: f32 [Mtiles*16, K] row-major (lda). B: packed bf16 fragments (NT n-tiles).
// One wave computes one 16x16 output tile; K loop of 16x16x32 bf16 WMMAs.
// Main loop is branch-free (full 32-wide k-tiles, 16B-aligned rows); a tail
// loop handles ragged K / unaligned lda (only the K=257 pair-feature GEMM).
__global__ __launch_bounds__(256) void wmma_gemm_kernel(
    const float* __restrict__ A, const unsigned short* __restrict__ Bp,
    const float* __restrict__ bias, float* __restrict__ C,
    int Mtiles, int K, int Nout, int NT, int lda, int ldc, int act)
{
  int gwid = (blockIdx.x * 256 + threadIdx.x) >> 5;
  int lane = threadIdx.x & 31;
  if (gwid >= Mtiles * NT) return;
  int mt = gwid / NT;
  int nt = gwid % NT;
  int half = lane >> 4;   // which K-half of the fragment this lane holds
  int mr   = lane & 15;
  const float* arow = A + (size_t)(mt * 16 + mr) * lda;
  int KT = (K + 31) >> 5;
  int KTfast = ((lda & 3) == 0) ? (K >> 5) : 0;   // full, aligned k-tiles
  const unsigned short* bp = Bp + ((size_t)nt * 32 + lane) * 16;
  size_t bstep = (size_t)NT * 512;                // one full k-tile of fragments
  v8f acc = {};
  int kt = 0;
#pragma unroll 2
  for (; kt < KTfast; ++kt) {
    int k0 = kt * 32 + half * 8;
    const float4* p = (const float4*)(arow + k0);
    float4 f0 = p[0], f1 = p[1];
    const float4* q = (const float4*)(arow + k0 + 16);
    float4 f2 = q[0], f3 = q[1];
    __builtin_prefetch(arow + k0 + 32, 0, 3);     // next k-tile, WGP-near
    v16bf a;
    // A fragment: elems 0..7 -> K=k0..k0+7 ; elems 8..15 -> K=k0+16..k0+23
    a[0]=(__bf16)f0.x;  a[1]=(__bf16)f0.y;  a[2]=(__bf16)f0.z;  a[3]=(__bf16)f0.w;
    a[4]=(__bf16)f1.x;  a[5]=(__bf16)f1.y;  a[6]=(__bf16)f1.z;  a[7]=(__bf16)f1.w;
    a[8]=(__bf16)f2.x;  a[9]=(__bf16)f2.y;  a[10]=(__bf16)f2.z; a[11]=(__bf16)f2.w;
    a[12]=(__bf16)f3.x; a[13]=(__bf16)f3.y; a[14]=(__bf16)f3.z; a[15]=(__bf16)f3.w;
    v16bf b = *(const v16bf*)bp;
    bp += bstep;
    acc = __builtin_amdgcn_wmma_f32_16x16x32_bf16(
        false, a, false, b, (short)0, acc, false, false);
  }
  for (; kt < KT; ++kt) {                         // ragged / unaligned tail
    int k0 = kt * 32 + half * 8;
    v16bf a;
#pragma unroll
    for (int e = 0; e < 8; ++e) {
      int k = k0 + e;
      a[e] = (k < K) ? (__bf16)arow[k] : (__bf16)0.0f;
    }
#pragma unroll
    for (int e = 0; e < 8; ++e) {
      int k = k0 + 16 + e;
      a[8 + e] = (k < K) ? (__bf16)arow[k] : (__bf16)0.0f;
    }
    v16bf b = *(const v16bf*)bp;
    bp += bstep;
    acc = __builtin_amdgcn_wmma_f32_16x16x32_bf16(
        false, a, false, b, (short)0, acc, false, false);
  }
  // D layout: VGPR v, lane l -> row = v + 8*(l>>4), col = l&15
  int col = nt * 16 + mr;
  if (col < Nout) {
    float bv = bias ? bias[col] : 0.0f;
#pragma unroll
    for (int v = 0; v < 8; ++v) {
      float o = acc[v] + bv;
      if (act == 1) o = fmaxf(o, 0.0f);
      C[(size_t)(mt * 16 + v + half * 8) * ldc + col] = o;
    }
  }
}

// ---------------- CRF negative log-likelihood (one thread per batch) -------
__global__ __launch_bounds__(64) void crf_nll_kernel(
    const float* __restrict__ em, const int* __restrict__ tags,
    const int* __restrict__ mask, const float* __restrict__ start,
    const float* __restrict__ endv, const float* __restrict__ trans,
    float* __restrict__ out_loss)
{
  __shared__ float tr[cK * cK], st[cK], en[cK], red[64];
  int tid = threadIdx.x;
  if (tid < cK * cK) tr[tid] = trans[tid];
  if (tid < cK) { st[tid] = start[tid]; en[tid] = endv[tid]; }
  __syncthreads();
  int b = tid;
  const float* emb = em + (size_t)b * cS * cK;
  float a[cK];
  int t0 = tags[b * cS];
  float num = st[t0] + emb[t0];
#pragma unroll
  for (int k = 0; k < cK; ++k) a[k] = st[k] + emb[k];
  int ptag = t0;
  for (int s = 1; s < cS; ++s) {
    const float* e = emb + s * cK;
    float ev[cK];
#pragma unroll
    for (int k = 0; k < cK; ++k) ev[k] = e[k];
    int m = mask[b * cS + s];
    int t = tags[b * cS + s];
    if (m) { num += tr[ptag * cK + t] + ev[t]; ptag = t; }
    float nxt[cK];
#pragma unroll
    for (int j = 0; j < cK; ++j) {
      float mx = -1e30f;
#pragma unroll
      for (int i = 0; i < cK; ++i) mx = fmaxf(mx, a[i] + tr[i * cK + j]);
      float sm = 0.f;
#pragma unroll
      for (int i = 0; i < cK; ++i) sm += expf(a[i] + tr[i * cK + j] - mx);
      nxt[j] = mx + logf(sm) + ev[j];
    }
    if (m) {
#pragma unroll
      for (int j = 0; j < cK; ++j) a[j] = nxt[j];
    }
  }
  num += en[ptag];
  float mx = -1e30f;
#pragma unroll
  for (int k = 0; k < cK; ++k) mx = fmaxf(mx, a[k] + en[k]);
  float sm = 0.f;
#pragma unroll
  for (int k = 0; k < cK; ++k) sm += expf(a[k] + en[k] - mx);
  float Z = mx + logf(sm);
  red[tid] = Z - num;
  __syncthreads();
  for (int o = 32; o > 0; o >>= 1) {
    if (tid < o) red[tid] += red[tid + o];
    __syncthreads();
  }
  if (tid == 0) out_loss[0] = red[0] / (float)cB;
}

// ---------------- entity span mean pooling ---------------------------------
__global__ __launch_bounds__(256) void entity_pool_kernel(
    const float* __restrict__ X, const int* __restrict__ estart,
    const int* __restrict__ elen, float* __restrict__ ent)
{
  int be = blockIdx.x;                 // b*E + e
  int s0 = estart[be];
  int ln = elen[be];
  int se = s0 + ln; if (se > cS - 1) se = cS - 1;
  float inv = 1.0f / (float)(se - s0 + 1);
  int b = be / cE;
  for (int h = threadIdx.x; h < cH; h += 256) {
    float sum = 0.f;
    for (int s = s0; s <= se; ++s)
      sum += X[((size_t)b * cS + s) * cH + h];
    ent[(size_t)be * cH + h] = sum * inv;
  }
}

// ---------------- dense GAT attention (per block: one (b, head)) -----------
__global__ __launch_bounds__(256) void gat_attn_kernel(
    const float* __restrict__ Hb, const unsigned char* __restrict__ adj,
    const float* __restrict__ a_s, const float* __restrict__ a_d,
    const float* __restrict__ bias, float* __restrict__ out,
    int heads, int cout, int relu_flag)
{
  extern __shared__ float sm[];
  float* hs  = sm;                       // E*cout   staged head slice
  float* sv  = hs + cE * cout;           // E
  float* dv  = sv + cE;                  // E
  float* att = dv + cE;                  // E*E
  int b  = blockIdx.x / heads;
  int hd = blockIdx.x % heads;
  int tid = threadIdx.x;
  int hcols = heads * cout;
  for (int idx = tid; idx < cE * cout; idx += 256) {
    int e = idx / cout, c = idx % cout;
    hs[idx] = Hb[((size_t)b * cE + e) * hcols + hd * cout + c];
  }
  __syncthreads();
  if (tid < 2 * cE) {
    int e = tid & (cE - 1);
    const float* av = (tid < cE) ? (a_s + hd * cout) : (a_d + hd * cout);
    float acc = 0.f;
    for (int c = 0; c < cout; ++c) acc += hs[e * cout + c] * av[c];
    if (tid < cE) sv[e] = acc; else dv[e] = acc;
  }
  __syncthreads();
  if (tid < cE) {
    int i = tid;
    float lg[cE];
    float mx = -1e30f;
#pragma unroll
    for (int j = 0; j < cE; ++j) {
      float l = dv[i] + sv[j];
      l = (l < 0.f) ? 0.2f * l : l;                       // leaky 0.2
      if (!adj[((size_t)b * cE + i) * cE + j]) l = -1e9f; // mask
      lg[j] = l; mx = fmaxf(mx, l);
    }
    float ssum = 0.f;
#pragma unroll
    for (int j = 0; j < cE; ++j) { float ex = expf(lg[j] - mx); lg[j] = ex; ssum += ex; }
    float inv = 1.f / ssum;
#pragma unroll
    for (int j = 0; j < cE; ++j) att[i * cE + j] = lg[j] * inv;
  }
  __syncthreads();
  for (int idx = tid; idx < cE * cout; idx += 256) {
    int i = idx / cout, c = idx % cout;
    float acc = 0.f;
#pragma unroll
    for (int j = 0; j < cE; ++j) acc += att[i * cE + j] * hs[j * cout + c];
    float o = acc + bias[hd * cout + c];
    if (relu_flag) o = fmaxf(o, 0.f);
    out[((size_t)b * cE + i) * hcols + hd * cout + c] = o;
  }
}

// ---------------- pair features: [x1, x2, x1*x2, ctx, logdist] -------------
__global__ __launch_bounds__(256) void pair_feat_kernel(
    const float* __restrict__ x, const int* __restrict__ pe1,
    const int* __restrict__ pe2, float* __restrict__ pf)
{
  __shared__ float xs[cE * 64];
  __shared__ float ctx[64];
  int b = blockIdx.x;
  int tid = threadIdx.x;
  for (int idx = tid; idx < cE * 64; idx += 256)
    xs[idx] = x[(size_t)b * cE * 64 + idx];
  __syncthreads();
  if (tid < 64) {
    float s = 0.f;
#pragma unroll
    for (int e = 0; e < cE; ++e) s += xs[e * 64 + tid];
    ctx[tid] = s * (1.f / (float)cE);
  }
  __syncthreads();
  for (int idx = tid; idx < cP * 64; idx += 256) {
    int p = idx / 64, c = idx % 64;
    int e1 = pe1[b * cP + p], e2 = pe2[b * cP + p];
    float x1v = xs[e1 * 64 + c], x2v = xs[e2 * 64 + c];
    float* row = pf + ((size_t)b * cP + p) * 257;
    row[c]        = x1v;
    row[64 + c]   = x2v;
    row[128 + c]  = x1v * x2v;
    row[192 + c]  = ctx[c];
    if (c == 0) row[256] = logf(fabsf((float)(e1 - e2)) + 1.0f);
  }
}

// ---------------- layernorm + leaky-relu (in place, one row per block) -----
__global__ __launch_bounds__(256) void ln_leaky_kernel(
    float* __restrict__ buf, const float* __restrict__ g,
    const float* __restrict__ be, int dim, int rows_per_group, float slope)
{
  __shared__ float red[256];
  int row = blockIdx.x;
  float* x = buf + (size_t)row * dim;
  int goff = (row / rows_per_group) * dim;
  int tid = threadIdx.x;
  float s = 0.f;
  for (int c = tid; c < dim; c += 256) s += x[c];
  red[tid] = s; __syncthreads();
  for (int o = 128; o > 0; o >>= 1) { if (tid < o) red[tid] += red[tid + o]; __syncthreads(); }
  float mean = red[0] / (float)dim;
  __syncthreads();
  float sq = 0.f;
  for (int c = tid; c < dim; c += 256) { float d = x[c] - mean; sq += d * d; }
  red[tid] = sq; __syncthreads();
  for (int o = 128; o > 0; o >>= 1) { if (tid < o) red[tid] += red[tid + o]; __syncthreads(); }
  float inv = rsqrtf(red[0] / (float)dim + 1e-5f);
  for (int c = tid; c < dim; c += 256) {
    float y = (x[c] - mean) * inv * g[goff + c] + be[goff + c];
    x[c] = (y < 0.f) ? slope * y : y;
  }
}

// ---------------- relation head final projection: z[r,b,p] -----------------
__global__ __launch_bounds__(256) void rc2_kernel(
    const float* __restrict__ h1, const float* __restrict__ w2,
    const float* __restrict__ b2, float* __restrict__ zout)
{
  int idx = blockIdx.x * 256 + threadIdx.x;   // R*B*P = 6144
  if (idx >= cR * cB * cP) return;
  int r = idx / (cB * cP);
  const float* hrow = h1 + (size_t)idx * 256;
  const float* w = w2 + r * 256;
  float acc = b2[r];
  for (int o = 0; o < 256; ++o) acc += hrow[o] * w[o];
  zout[idx] = acc;
}

// ---------------- total loss: ner_loss + sum_r mean(bce) -------------------
__global__ __launch_bounds__(256) void final_loss_kernel(
    const float* __restrict__ z, const int* __restrict__ ylab,
    const float* __restrict__ ner_loss, float* __restrict__ out_total)
{
  __shared__ float red[256];
  int tid = threadIdx.x;
  float s = 0.f;
  for (int idx = tid; idx < cR * cB * cP; idx += 256) {
    int bp = idx % (cB * cP);
    float y  = (float)ylab[bp];
    float zv = z[idx];
    float l1 = log1pf(expf(-fabsf(zv)));
    float ls_p = fminf(zv, 0.f) - l1;    // log sigmoid(z)
    float ls_n = -fmaxf(zv, 0.f) - l1;   // log sigmoid(-z)
    s += -(2.0f * y * ls_p + (1.0f - y) * ls_n);
  }
  red[tid] = s; __syncthreads();
  for (int o = 128; o > 0; o >>= 1) { if (tid < o) red[tid] += red[tid + o]; __syncthreads(); }
  if (tid == 0) out_total[0] = ner_loss[0] + red[0] / (float)(cB * cP);
}

// ===========================================================================
extern "C" void kernel_launch(void* const* d_in, const int* in_sizes, int n_in,
                              void* d_out, int out_size, void* d_ws, size_t ws_size,
                              hipStream_t stream) {
  (void)in_sizes; (void)n_in; (void)out_size; (void)ws_size;
  const float* seq    = (const float*)d_in[0];
  const int*   amask  = (const int*)d_in[1];
  const int*   tags   = (const int*)d_in[2];
  const int*   est    = (const int*)d_in[3];
  const int*   eln    = (const int*)d_in[4];
  const int*   pe1    = (const int*)d_in[5];
  const int*   pe2    = (const int*)d_in[6];
  const int*   plab   = (const int*)d_in[7];
  const unsigned char* adj = (const unsigned char*)d_in[8];
  const float* w_ner1 = (const float*)d_in[9];
  const float* b_ner1 = (const float*)d_in[10];
  const float* w_ner2 = (const float*)d_in[11];
  const float* b_ner2 = (const float*)d_in[12];
  const float* crf_s  = (const float*)d_in[13];
  const float* crf_e  = (const float*)d_in[14];
  const float* crf_t  = (const float*)d_in[15];
  const float* w_gat1 = (const float*)d_in[16];
  const float* a_src1 = (const float*)d_in[17];
  const float* a_dst1 = (const float*)d_in[18];
  const float* b_gat1 = (const float*)d_in[19];
  const float* w_gat2 = (const float*)d_in[20];
  const float* a_src2 = (const float*)d_in[21];
  const float* a_dst2 = (const float*)d_in[22];
  const float* b_gat2 = (const float*)d_in[23];
  const float* w_fe   = (const float*)d_in[24];
  const float* b_fe   = (const float*)d_in[25];
  const float* g_fe   = (const float*)d_in[26];
  const float* be_fe  = (const float*)d_in[27];
  const float* w_rc1  = (const float*)d_in[28];
  const float* b_rc1  = (const float*)d_in[29];
  const float* g_rc   = (const float*)d_in[30];
  const float* be_rc  = (const float*)d_in[31];
  const float* w_rc2  = (const float*)d_in[32];
  const float* b_rc2  = (const float*)d_in[33];

  // workspace carving (256B aligned)
  char* ws = (char*)d_ws;
  size_t off = 0;
  auto take = [&](size_t bytes) -> char* {
    char* p = ws + off;
    off += (bytes + 255) & ~(size_t)255;
    return p;
  };
  unsigned short* pk_ner1 = (unsigned short*)take((size_t)24 * 16 * 512 * 2);
  unsigned short* pk_ner2 = (unsigned short*)take((size_t)8  *  1 * 512 * 2);
  unsigned short* pk_gat1 = (unsigned short*)take((size_t)24 * 32 * 512 * 2);
  unsigned short* pk_gat2 = (unsigned short*)take((size_t)16 *  4 * 512 * 2);
  unsigned short* pk_fe   = (unsigned short*)take((size_t)9  * 32 * 512 * 2);
  unsigned short* pk_rc1  = (unsigned short*)take((size_t)6 * 16 * 16 * 512 * 2);
  float* hid   = (float*)take((size_t)cB * cS * 256 * 4);
  float* ent   = (float*)take((size_t)cB * cE * cH * 4);
  float* h1b   = (float*)take((size_t)cB * cE * 512 * 4);
  float* xbuf  = (float*)take((size_t)cB * cE * 512 * 4);
  float* h2b   = (float*)take((size_t)cB * cE * 64 * 4);
  float* x2b   = (float*)take((size_t)cB * cE * 64 * 4);
  float* pf    = (float*)take((size_t)cB * cP * 257 * 4);
  float* febuf = (float*)take((size_t)cB * cP * 512 * 4);
  float* h1r   = (float*)take((size_t)cR * cB * cP * 256 * 4);
  float* nloss = (float*)take(256);

  float* ner_logits = (float*)d_out;                          // [B,S,9]
  float* zout  = (float*)d_out + (size_t)cB * cS * cK;        // [R,B,P]
  float* total = zout + (size_t)cR * cB * cP;                 // scalar

  // --- pack all GEMM B operands into bf16 WMMA fragments ---
  pack_b_kernel<<<48, 256, 0, stream>>>(w_ner1, pk_ner1, 768, 256, 16);
  pack_b_kernel<<<1,  256, 0, stream>>>(w_ner2, pk_ner2, 256, 9,   1);
  pack_b_kernel<<<96, 256, 0, stream>>>(w_gat1, pk_gat1, 768, 512, 32);
  pack_b_kernel<<<8,  256, 0, stream>>>(w_gat2, pk_gat2, 512, 64,  4);
  pack_b_kernel<<<36, 256, 0, stream>>>(w_fe,   pk_fe,   257, 512, 32);
  for (int r = 0; r < cR; ++r)
    pack_b_kernel<<<32, 256, 0, stream>>>(w_rc1 + (size_t)r * 512 * 256,
                                          pk_rc1 + (size_t)r * 16 * 16 * 512,
                                          512, 256, 16);

  // --- NER head: hid = relu(X @ W1 + b1); logits = hid @ W2 + b2 ---
  wmma_gemm_kernel<<<4096, 256, 0, stream>>>(seq, pk_ner1, b_ner1, hid,
                                             2048, 768, 256, 16, 768, 256, 1);
  wmma_gemm_kernel<<<256, 256, 0, stream>>>(hid, pk_ner2, b_ner2, ner_logits,
                                            2048, 256, 9, 1, 256, 9, 0);
  crf_nll_kernel<<<1, 64, 0, stream>>>(ner_logits, tags, amask,
                                       crf_s, crf_e, crf_t, nloss);

  // --- entity pooling + 2-layer GAT ---
  entity_pool_kernel<<<cB * cE, 256, 0, stream>>>(seq, est, eln, ent);
  wmma_gemm_kernel<<<512, 256, 0, stream>>>(ent, pk_gat1, nullptr, h1b,
                                            128, 768, 512, 32, 768, 512, 0);
  gat_attn_kernel<<<cB * 4, 256, (cE * 128 + 2 * cE + cE * cE) * 4, stream>>>(
      h1b, adj, a_src1, a_dst1, b_gat1, xbuf, 4, 128, 1);
  wmma_gemm_kernel<<<64, 256, 0, stream>>>(xbuf, pk_gat2, nullptr, h2b,
                                           128, 512, 64, 4, 512, 64, 0);
  gat_attn_kernel<<<cB, 256, (cE * 64 + 2 * cE + cE * cE) * 4, stream>>>(
      h2b, adj, a_src2, a_dst2, b_gat2, x2b, 1, 64, 0);

  // --- pair features -> fused FC + LN + relation classifiers ---
  pair_feat_kernel<<<cB, 256, 0, stream>>>(x2b, pe1, pe2, pf);
  wmma_gemm_kernel<<<256, 256, 0, stream>>>(pf, pk_fe, b_fe, febuf,
                                            64, 257, 512, 32, 257, 512, 0);
  ln_leaky_kernel<<<cB * cP, 256, 0, stream>>>(febuf, g_fe, be_fe, 512, cB * cP, 0.01f);
  for (int r = 0; r < cR; ++r)
    wmma_gemm_kernel<<<128, 256, 0, stream>>>(febuf,
        pk_rc1 + (size_t)r * 16 * 16 * 512, b_rc1 + r * 256,
        h1r + (size_t)r * cB * cP * 256, 64, 512, 256, 16, 512, 256, 0);
  ln_leaky_kernel<<<cR * cB * cP, 256, 0, stream>>>(h1r, g_rc, be_rc, 256, cB * cP, 0.01f);
  rc2_kernel<<<24, 256, 0, stream>>>(h1r, w_rc2, b_rc2, zout);
  final_loss_kernel<<<1, 256, 0, stream>>>(zout, plab, nloss, total);
}
